// MultiHeadAttention_11003706213310
// MI455X (gfx1250) — compile-verified, matrix operand order checked by reading, not run on hardware
//
#include <hip/hip_runtime.h>
#include <hip/hip_bf16.h>

// ---------------------------------------------------------------------------
// MHA forward for MI455X (gfx1250, wave32, WMMA).
// Round 3: grow per-wave tiles to rebalance the loop toward the matrix cores.
//  - GEMMs: 32x64 per wave (2 A-frags x 4 B-frags -> 8 WMMA / 12 loads)
//  - Attention: 32 query rows per wave; K/V fragments shared by both 16-row
//    halves (16 WMMA / 16 loads per 32-key step); causal branch eliminated.
// All operands pre-converted to f16 once; hot loops are loads+WMMA only.
// ---------------------------------------------------------------------------

typedef __attribute__((ext_vector_type(16))) _Float16 v16h;
typedef __attribute__((ext_vector_type(8)))  _Float16 v8h;
typedef __attribute__((ext_vector_type(8)))  float    v8f;

#define S_LEN   4096
#define DMODEL  768
#define NHEADS  12
#define DK      64
#define PB_STRIDE 40   // LDS row stride in halves: 80B => 16B-aligned b128 reads, conflict-free

// A/B fragment element mapping for v_wmma_f32_16x16x32_f16 (wave32):
// lane l: row/col = l&15, group g = l>>4; the 16 halves cover
// K = {g*8 .. g*8+7} and {16+g*8 .. 16+g*8+7}  (two contiguous runs of 8).

static __device__ __forceinline__ v16h ldfrag_h(const _Float16* p) {
  v8h lo = *(const v8h*)(p);        // run0: 16B
  v8h hi = *(const v8h*)(p + 16);   // run1: 16B
  v16h r;
#pragma unroll
  for (int i = 0; i < 8; ++i) { r[i] = lo[i]; r[i + 8] = hi[i]; }
  return r;
}

static __device__ __forceinline__ v8f wmma_f16(v16h a, v16h b, v8f c) {
  return __builtin_amdgcn_wmma_f32_16x16x32_f16(false, a, false, b, (short)0, c,
                                                false, false);
}

// ---------------------------------------------------------------------------
// One-pass f32 -> f16 conversion, 8 elements/thread.
// ---------------------------------------------------------------------------
__global__ __launch_bounds__(256) void cvt_f16_kernel(
    const float* __restrict__ src, _Float16* __restrict__ dst, int n8) {
  const int i = blockIdx.x * blockDim.x + threadIdx.x;
  if (i >= n8) return;
  float4 a = ((const float4*)src)[2 * i];
  float4 b = ((const float4*)src)[2 * i + 1];
  v8h r;
  r[0] = (_Float16)a.x; r[1] = (_Float16)a.y; r[2] = (_Float16)a.z; r[3] = (_Float16)a.w;
  r[4] = (_Float16)b.x; r[5] = (_Float16)b.y; r[6] = (_Float16)b.z; r[7] = (_Float16)b.w;
  ((v8h*)dst)[i] = r;
}

// ---------------------------------------------------------------------------
// Projection: Y[s,d] = sum_k X[s,k]*W[d,k] + B[d]   (y = x @ W^T + b)
// One wave: 32x64 tile. B fragments shared by two 16-row A fragments.
// vt_mode==0: write f16 [h][s][dk] (Q,K).  vt_mode==1: write f16 [h][dk][s]
// (V transposed; D-frag rows contiguous there => packed 16B stores).
// ---------------------------------------------------------------------------
__global__ __launch_bounds__(256) void proj_kernel(
    const _Float16* __restrict__ X, const _Float16* __restrict__ W,
    const float* __restrict__ Bv, _Float16* __restrict__ Y, int vt_mode) {
  const int wid  = (blockIdx.x * blockDim.x + threadIdx.x) >> 5;
  const int lane = threadIdx.x & 31;
  const int g = lane >> 4, n16 = lane & 15;
  const int mtile = wid % (S_LEN / 32);
  const int ntile = wid / (S_LEN / 32);
  const int rowBase = mtile * 32;
  const int colBase = ntile * 64;

  v8f acc[2][4] = {};
  const _Float16* xrow0 = X + (size_t)(rowBase + n16) * DMODEL;
  const _Float16* xrow1 = xrow0 + (size_t)16 * DMODEL;

  for (int kk = 0; kk < DMODEL; kk += 32) {
    v16h a0 = ldfrag_h(xrow0 + kk + g * 8);
    v16h a1 = ldfrag_h(xrow1 + kk + g * 8);
#pragma unroll
    for (int t = 0; t < 4; ++t) {
      const _Float16* wrow =
          W + (size_t)(colBase + t * 16 + n16) * DMODEL + kk + g * 8;
      v16h b = ldfrag_h(wrow);
      acc[0][t] = wmma_f16(a0, b, acc[0][t]);
      acc[1][t] = wmma_f16(a1, b, acc[1][t]);
    }
  }
#pragma unroll
  for (int t = 0; t < 4; ++t) {
    const int dm = colBase + t * 16 + n16;       // output column (fixed per lane)
    const float bb = Bv[dm];
    const int h = dm >> 6, dk = dm & 63;
#pragma unroll
    for (int half = 0; half < 2; ++half) {
      const int rb = rowBase + half * 16 + 8 * g;
      if (vt_mode == 0) {
        _Float16* yb = Y + ((size_t)h * S_LEN + rb) * DK + dk;
#pragma unroll
        for (int r = 0; r < 8; ++r) yb[(size_t)r * DK] = (_Float16)(acc[half][t][r] + bb);
      } else {
        _Float16* yb = Y + ((size_t)(h * DK + dk)) * S_LEN + rb;
        v8h pk;
#pragma unroll
        for (int r = 0; r < 8; ++r) pk[r] = (_Float16)(acc[half][t][r] + bb);
        *(v8h*)yb = pk;                           // 16B packed store
      }
    }
  }
}

// ---------------------------------------------------------------------------
// Flash attention (causal), one wave per (head, 32-query tile).
// K/V fragments loaded once per 32-key step and shared by both 16-row halves.
// Online softmax via shfl_xor row reductions; P transposed C->A layout via a
// wave-private LDS slot (DS ops in-order per wave => no barriers).
// ---------------------------------------------------------------------------
__global__ __launch_bounds__(256) void attn_kernel(
    const _Float16* __restrict__ Qh, const _Float16* __restrict__ Kh,
    const _Float16* __restrict__ Vt, _Float16* __restrict__ Oh) {
  __shared__ _Float16 pbuf[8][32 * PB_STRIDE];
  const int wslot = threadIdx.x >> 5;
  const int wid   = (blockIdx.x * blockDim.x + threadIdx.x) >> 5;
  const int lane  = threadIdx.x & 31;
  const int g = lane >> 4, n16 = lane & 15;
  const int qtile = wid % (S_LEN / 32);
  const int h     = wid / (S_LEN / 32);
  const int rowBase = qtile * 32;

  const _Float16* qbA = Qh + ((size_t)h * S_LEN + rowBase + n16) * DK;
  const _Float16* qbB = qbA + (size_t)16 * DK;
  v16h aq[2][2];
  aq[0][0] = ldfrag_h(qbA + g * 8);        // rows 0..15,  K 0..31
  aq[0][1] = ldfrag_h(qbA + 32 + g * 8);   // rows 0..15,  K 32..63
  aq[1][0] = ldfrag_h(qbB + g * 8);        // rows 16..31, K 0..31
  aq[1][1] = ldfrag_h(qbB + 32 + g * 8);   // rows 16..31, K 32..63

  v8f o[2][4] = {};
  float mrow[2][8], lrow[2][8];
#pragma unroll
  for (int hh = 0; hh < 2; ++hh)
#pragma unroll
    for (int r = 0; r < 8; ++r) { mrow[hh][r] = -1e30f; lrow[hh][r] = 0.0f; }

  _Float16* pb = &pbuf[wslot][0];

  // kb runs over multiples of 32 up to rowBase (diagonal block pair included);
  // both 16-key tiles of each step are causally reachable for the 32-row tile.
  for (int kb = 0; kb <= rowBase; kb += 32) {
    // K fragments for both 16-key tiles (shared by both query halves)
    const _Float16* kp0 = Kh + ((size_t)h * S_LEN + kb + n16) * DK;
    const _Float16* kp1 = kp0 + (size_t)16 * DK;
    v16h bk00 = ldfrag_h(kp0 + g * 8);
    v16h bk01 = ldfrag_h(kp0 + 32 + g * 8);
    v16h bk10 = ldfrag_h(kp1 + g * 8);
    v16h bk11 = ldfrag_h(kp1 + 32 + g * 8);

    v8f sc[2][2];
#pragma unroll
    for (int hh = 0; hh < 2; ++hh) {
      v8f c0 = {};
      c0 = wmma_f16(aq[hh][0], bk00, c0);
      c0 = wmma_f16(aq[hh][1], bk01, c0);
      sc[hh][0] = c0;
      v8f c1 = {};
      c1 = wmma_f16(aq[hh][0], bk10, c1);
      c1 = wmma_f16(aq[hh][1], bk11, c1);
      sc[hh][1] = c1;
    }
    __builtin_prefetch((const void*)(Kh + ((size_t)h * S_LEN + kb + 32) * DK), 0, 1);

#pragma unroll
    for (int hh = 0; hh < 2; ++hh) {
      // scale by 1/sqrt(dk) and causal mask (value select, no divergence)
#pragma unroll
      for (int t = 0; t < 2; ++t) {
        const int kj = kb + t * 16 + n16;          // key index (fixed per lane)
#pragma unroll
        for (int r = 0; r < 8; ++r) {
          const int qi = rowBase + hh * 16 + r + 8 * g;   // query row
          const float s = sc[hh][t][r] * 0.125f;
          sc[hh][t][r] = (kj > qi) ? -1e30f : s;
        }
      }
      // online softmax per row; each row lives across one 16-lane half
#pragma unroll
      for (int r = 0; r < 8; ++r) {
        float mx = fmaxf(sc[hh][0][r], sc[hh][1][r]);
#pragma unroll
        for (int off = 1; off < 16; off <<= 1) mx = fmaxf(mx, __shfl_xor(mx, off, 32));
        const float mnew  = fmaxf(mrow[hh][r], mx);
        const float alpha = __expf(mrow[hh][r] - mnew);
        const float p0 = __expf(sc[hh][0][r] - mnew);
        const float p1 = __expf(sc[hh][1][r] - mnew);
        float rs = p0 + p1;
#pragma unroll
        for (int off = 1; off < 16; off <<= 1) rs += __shfl_xor(rs, off, 32);
        lrow[hh][r] = lrow[hh][r] * alpha + rs;
        mrow[hh][r] = mnew;
#pragma unroll
        for (int t2 = 0; t2 < 4; ++t2) o[hh][t2][r] *= alpha;
        // C-layout -> LDS (row = hh*16 + r + 8g, cols n16 and n16+16)
        pb[(hh * 16 + r + 8 * g) * PB_STRIDE + n16]      = (_Float16)p0;
        pb[(hh * 16 + r + 8 * g) * PB_STRIDE + 16 + n16] = (_Float16)p1;
      }
    }

    // LDS -> A-fragments of P (16x32 each): row m = n16 (+16), K runs {g*8, 16+g*8}
    v16h pa0 = ldfrag_h(pb + n16 * PB_STRIDE + g * 8);
    v16h pa1 = ldfrag_h(pb + (16 + n16) * PB_STRIDE + g * 8);

    // O += P @ V  (V transposed: contiguous 16B runs; shared by both halves)
#pragma unroll
    for (int t2 = 0; t2 < 4; ++t2) {
      const _Float16* vp =
          Vt + ((size_t)(h * DK + t2 * 16 + n16)) * S_LEN + kb + g * 8;
      v16h bv = ldfrag_h(vp);
      o[0][t2] = wmma_f16(pa0, bv, o[0][t2]);
      o[1][t2] = wmma_f16(pa1, bv, o[1][t2]);
    }
  }

#pragma unroll
  for (int hh = 0; hh < 2; ++hh) {
    float invl[8];
#pragma unroll
    for (int r = 0; r < 8; ++r) invl[r] = 1.0f / lrow[hh][r];
#pragma unroll
    for (int t2 = 0; t2 < 4; ++t2) {
      const int dk = t2 * 16 + n16;
      _Float16* ob =
          Oh + ((size_t)h * S_LEN + rowBase + hh * 16 + 8 * g) * DK + dk;
#pragma unroll
      for (int r = 0; r < 8; ++r)
        ob[(size_t)r * DK] = (_Float16)(o[hh][t2][r] * invl[r]);
    }
  }
}

// ---------------------------------------------------------------------------
// Output projection: out[s,d] = sum_k O[s,k]*Wo[d,k] + bo[d], fp32 output.
// 32x64 per-wave tile; all-f16 fragments. Runs of 8 stay in-head.
// ---------------------------------------------------------------------------
__global__ __launch_bounds__(256) void outproj_kernel(
    const _Float16* __restrict__ Oh, const _Float16* __restrict__ WOh,
    const float* __restrict__ bo, float* __restrict__ out) {
  const int wid  = (blockIdx.x * blockDim.x + threadIdx.x) >> 5;
  const int lane = threadIdx.x & 31;
  const int g = lane >> 4, n16 = lane & 15;
  const int mtile = wid % (S_LEN / 32);
  const int ntile = wid / (S_LEN / 32);
  const int rowBase = mtile * 32;
  const int colBase = ntile * 64;

  v8f acc[2][4] = {};
  for (int kk = 0; kk < DMODEL; kk += 32) {
    const int k0 = kk + g * 8;       // run bases (multiples of 8, in-head)
    const int k1 = k0 + 16;
    const size_t off0 = (size_t)(k0 >> 6) * S_LEN * DK + (k0 & 63);
    const size_t off1 = (size_t)(k1 >> 6) * S_LEN * DK + (k1 & 63);
    v16h a[2];
#pragma unroll
    for (int half = 0; half < 2; ++half) {
      const size_t rowoff = (size_t)(rowBase + half * 16 + n16) * DK;
      v8h lo = *(const v8h*)(Oh + off0 + rowoff);
      v8h hi = *(const v8h*)(Oh + off1 + rowoff);
#pragma unroll
      for (int i = 0; i < 8; ++i) { a[half][i] = lo[i]; a[half][i + 8] = hi[i]; }
    }
#pragma unroll
    for (int t = 0; t < 4; ++t) {
      const _Float16* wrow =
          WOh + (size_t)(colBase + t * 16 + n16) * DMODEL + k0;
      v16h b = ldfrag_h(wrow);
      acc[0][t] = wmma_f16(a[0], b, acc[0][t]);
      acc[1][t] = wmma_f16(a[1], b, acc[1][t]);
    }
  }
#pragma unroll
  for (int t = 0; t < 4; ++t) {
    const int dm = colBase + t * 16 + n16;
    const float bb = bo[dm];
#pragma unroll
    for (int half = 0; half < 2; ++half) {
      float* ob = out + (size_t)(rowBase + half * 16 + 8 * g) * DMODEL + dm;
#pragma unroll
      for (int r = 0; r < 8; ++r) ob[(size_t)r * DMODEL] = acc[half][t][r] + bb;
    }
  }
}

// ---------------------------------------------------------------------------
extern "C" void kernel_launch(void* const* d_in, const int* in_sizes, int n_in,
                              void* d_out, int out_size, void* d_ws, size_t ws_size,
                              hipStream_t stream) {
  const float* q  = (const float*)d_in[0];
  const float* k  = (const float*)d_in[1];
  const float* v  = (const float*)d_in[2];
  // d_in[3] = causal mask (int32) — applied analytically in attn_kernel
  const float* wq = (const float*)d_in[4];
  const float* bq = (const float*)d_in[5];
  const float* wk = (const float*)d_in[6];
  const float* bk = (const float*)d_in[7];
  const float* wv = (const float*)d_in[8];
  const float* bv = (const float*)d_in[9];
  const float* wo = (const float*)d_in[10];
  const float* bo = (const float*)d_in[11];
  float* out = (float*)d_out;

  const size_t per  = (size_t)NHEADS * S_LEN * DK;   // 3,145,728 halves (== S*DMODEL)
  const size_t wsz  = (size_t)DMODEL * DMODEL;       //   589,824 halves
  _Float16* Qh  = (_Float16*)d_ws;    // [h][s][dk]
  _Float16* Kh  = Qh  + per;          // [h][s][dk]
  _Float16* Vt  = Kh  + per;          // [h][dk][s]
  _Float16* Oh  = Vt  + per;          // [h][s][dk]
  _Float16* Xq  = Oh  + per;          // f16 copies of inputs
  _Float16* Xk  = Xq  + per;
  _Float16* Xv  = Xk  + per;
  _Float16* Wq  = Xv  + per;          // f16 copies of weights
  _Float16* Wk  = Wq  + wsz;
  _Float16* Wv  = Wk  + wsz;
  _Float16* Wo  = Wv  + wsz;

  dim3 blk(256);
  const int nX8 = (int)(per / 8), nW8 = (int)(wsz / 8);
  dim3 gX((nX8 + 255) / 256), gW((nW8 + 255) / 256);

  cvt_f16_kernel<<<gX, blk, 0, stream>>>(q,  Xq, nX8);
  cvt_f16_kernel<<<gX, blk, 0, stream>>>(k,  Xk, nX8);
  cvt_f16_kernel<<<gX, blk, 0, stream>>>(v,  Xv, nX8);
  cvt_f16_kernel<<<gW, blk, 0, stream>>>(wq, Wq, nW8);
  cvt_f16_kernel<<<gW, blk, 0, stream>>>(wk, Wk, nW8);
  cvt_f16_kernel<<<gW, blk, 0, stream>>>(wv, Wv, nW8);
  cvt_f16_kernel<<<gW, blk, 0, stream>>>(wo, Wo, nW8);

  dim3 gproj((S_LEN / 32) * (DMODEL / 64) / 8);    // 192 blocks, 8 waves each
  dim3 gattn((NHEADS * (S_LEN / 32)) / 8);         // 192 blocks

  proj_kernel<<<gproj, blk, 0, stream>>>(Xq, Wq, bq, Qh, 0);
  proj_kernel<<<gproj, blk, 0, stream>>>(Xk, Wk, bk, Kh, 0);
  proj_kernel<<<gproj, blk, 0, stream>>>(Xv, Wv, bv, Vt, 1);
  attn_kernel<<<gattn, blk, 0, stream>>>(Qh, Kh, Vt, Oh);
  outproj_kernel<<<gproj, blk, 0, stream>>>(Oh, Wo, bo, out);
}
